// SetGNN_17343077941259
// MI455X (gfx1250) — matrix operand order repaired and checked by reading, not compiled, not run on hardware
//
#include <hip/hip_runtime.h>
#include <hip/hip_bf16.h>

#define NN  200000
#define NE  2000000
#define NHE 100000
#define FD  128

typedef __attribute__((ext_vector_type(16))) __bf16 v16bf;
typedef __attribute__((ext_vector_type(8)))  __bf16 v8bf;
typedef __attribute__((ext_vector_type(8)))  float  v8f;
typedef __attribute__((ext_vector_type(4)))  int    v4i;

#define KTR 136  // padded k-stride (halfs): 272 B/row -> 16B-aligned, banks rotate by 4
#define WT_BYTES (128 * KTR * 2)  // one transposed bf16 weight tile: 34816 B

// ---------------- async global->LDS staging (CDNA5), with safe fallback ------
#if defined(__has_builtin)
#if __has_builtin(__builtin_amdgcn_global_load_async_to_lds_b128) && \
    __has_builtin(__builtin_amdgcn_s_wait_asynccnt)
#define USE_ASYNC_LDS 1
#endif
#endif
#ifndef USE_ASYNC_LDS
#define USE_ASYNC_LDS 0
#endif

typedef __attribute__((address_space(1))) v4i* gbl_v4i_p;  // global int4*
typedef __attribute__((address_space(3))) v4i* lds_v4i_p;  // LDS int4*

__device__ __forceinline__ void lds_stage16(void* l, const void* g) {
#if USE_ASYNC_LDS
  __builtin_amdgcn_global_load_async_to_lds_b128((gbl_v4i_p)g, (lds_v4i_p)l, 0, 0);
#else
  *(float4*)l = *(const float4*)g;
#endif
}
__device__ __forceinline__ void lds_stage_wait() {
#if USE_ASYNC_LDS
  __builtin_amdgcn_s_wait_asynccnt(0);
#endif
}

// ---------------- WMMA helpers (CDNA5 layouts, CDNA5 ISA §7.12.2) --

__device__ __forceinline__ v8f wmma_bf(v16bf a, v16bf b, v8f c) {
  // D = A(16x32 bf16) x B(32x16 bf16) + C(16x16 f32)
  return __builtin_amdgcn_wmma_f32_16x16x32_bf16(false, a, false, b, (short)0, c, false, false);
}

__device__ __forceinline__ v16bf cat8(v8bf x0, v8bf x1) {
  v16bf r;
#pragma unroll
  for (int e = 0; e < 8; ++e) { r[e] = x0[e]; r[e + 8] = x1[e]; }
  return r;
}

// A fragment from row-major bf16 tile (two contiguous 16B runs)
__device__ __forceinline__ v16bf afrag_b(const __bf16* s, int stride, int m, int kc, int hi) {
  const __bf16* p = s + m * stride + kc * 32 + hi * 8;
  return cat8(*(const v8bf*)(p), *(const v8bf*)(p + 16));
}

// B fragment from TRANSPOSED W tile sWT[n][k]: two 16B runs (k=8hi.., 16+8hi..) -> 2x ds_load_b128
__device__ __forceinline__ v16bf bfrag_t(const __bf16* sWT, int n, int kc, int hi) {
  const __bf16* p = sWT + n * KTR + kc * 32 + hi * 8;
  return cat8(*(const v8bf*)(p), *(const v8bf*)(p + 16));
}

// A fragment straight from global fp32 rows (compiler merges to b128 loads)
__device__ __forceinline__ v16bf afrag_gf32(const float* xrow, int kc, int hi) {
  v16bf a;
#pragma unroll
  for (int e = 0; e < 16; ++e) {
    int v = e >> 1;
    int k = kc * 32 + ((v & 4) ? 16 : 0) + hi * 8 + (v & 3) * 2 + (e & 1);
    a[e] = (__bf16)xrow[k];
  }
  return a;
}

// ---------------- order-preserving float<->uint key for atomicMax ------------
__device__ __forceinline__ unsigned fenc(float f) {
  unsigned u = __float_as_uint(f);
  return (u & 0x80000000u) ? ~u : (u | 0x80000000u);
}
__device__ __forceinline__ float fdec(unsigned k) {
  return __uint_as_float((k & 0x80000000u) ? (k ^ 0x80000000u) : ~k);
}
#define ENC_NEGINF 0x007FFFFFu  // fenc(-inf)

// ---------------- K_pre: transpose+convert all weights to bf16 W^T tiles -----
__global__ void __launch_bounds__(256) k_pre(const float* __restrict__ WK,
                                             const float* __restrict__ WV,
                                             const float* __restrict__ W1,
                                             const float* __restrict__ W2,
                                             __bf16* __restrict__ wt) {
  int i = blockIdx.x * 256 + threadIdx.x;  // 4 * 16384 threads total
  int m = i >> 14, idx = i & 16383;
  int k = idx >> 7, n = idx & 127;  // W[k][n] -> W^T[n][k]
  const float* src = (m == 0) ? WK : (m == 1) ? WV : (m == 2) ? W1 : W2;
  wt[m * (128 * KTR) + n * KTR + k] = (__bf16)src[idx];
}

// ---------------- K0: init accumulators --------------------------------------
__global__ void __launch_bounds__(256) k0_init(unsigned* amaxU, float* denom, float* agg) {
  unsigned i = blockIdx.x * 256u + threadIdx.x;
  if (i < (unsigned)NHE * 4u) { amaxU[i] = ENC_NEGINF; denom[i] = 0.f; }
  if (i < (unsigned)NHE * 128u) agg[i] = 0.f;
}

// ---------------- K1: fused K/V projection + attention logits ----------------
__global__ void __launch_bounds__(256) k1_proj(const float* __restrict__ x,
                                               const __bf16* __restrict__ wkvT,  // W_K^T|W_V^T bf16
                                               const float* __restrict__ bK,
                                               const float* __restrict__ bV,
                                               const float* __restrict__ att,
                                               __bf16* __restrict__ xVb,
                                               float* __restrict__ aNode) {
  extern __shared__ unsigned char smem[];
  __bf16* sWKt = (__bf16*)smem;                  // [128][KTR]
  __bf16* sWVt = (__bf16*)(smem + WT_BYTES);     // [128][KTR]
  float*  sAtt = (float*)(smem + 2 * WT_BYTES);  // [128]
  const int tid = threadIdx.x;

  // async-DMA both transposed weight tiles (contiguous 2*WT_BYTES) into LDS
  for (int i = tid; i < (2 * WT_BYTES) / 16; i += 256)
    lds_stage16(smem + i * 16, (const char*)wkvT + i * 16);
  if (tid < 128) sAtt[tid] = att[tid];

  const int lane = tid & 31, wv = tid >> 5;
  const int lo = lane & 15, hi = lane >> 4;
  const int rowBase = blockIdx.x * 128 + wv * 16;
  int arow = rowBase + lo;
  if (arow >= NN) arow = NN - 1;
  const float* xrow = x + (size_t)arow * FD;

  // overlap the weight DMA with the global A-fragment loads
  v16bf aF[4];
#pragma unroll
  for (int kc = 0; kc < 4; ++kc) aF[kc] = afrag_gf32(xrow, kc, hi);

  lds_stage_wait();
  __syncthreads();

#pragma unroll
  for (int h = 0; h < 4; ++h) {
    v8f cK[2];
#pragma unroll
    for (int p = 0; p < 2; ++p) {
      const int t = 2 * h + p;
      v8f aK = {}, aV = {};
#pragma unroll
      for (int kc = 0; kc < 4; ++kc) {
        aK = wmma_bf(aF[kc], bfrag_t(sWKt, t * 16 + lo, kc, hi), aK);
        aV = wmma_bf(aF[kc], bfrag_t(sWVt, t * 16 + lo, kc, hi), aV);
      }
      cK[p] = aK;
      const float bv = bV[t * 16 + lo];
#pragma unroll
      for (int j = 0; j < 8; ++j) {
        int node = rowBase + j + 8 * hi;
        if (node < NN) xVb[(size_t)node * FD + t * 16 + lo] = (__bf16)(aV[j] + bv);
      }
    }
    // alpha[M][h] = sum_c (xK + bK)[c] * att[h][c], reduced across the 16-lane half
    const float attA = sAtt[h * 32 + lo],  attB = sAtt[h * 32 + 16 + lo];
    const float bk0  = bK[h * 32 + lo],    bk1  = bK[h * 32 + 16 + lo];
    float part[8];
#pragma unroll
    for (int j = 0; j < 8; ++j)
      part[j] = (cK[0][j] + bk0) * attA + (cK[1][j] + bk1) * attB;
#pragma unroll
    for (int msk = 8; msk >= 1; msk >>= 1)
#pragma unroll
      for (int j = 0; j < 8; ++j) part[j] += __shfl_xor(part[j], msk, 32);
    if (lo == h) {
#pragma unroll
      for (int j = 0; j < 8; ++j) {
        int node = rowBase + j + 8 * hi;
        if (node < NN) {
          float a = part[j];
          aNode[node * 4 + h] = a > 0.f ? a : 0.2f * a;  // leaky_relu folded in
        }
      }
    }
  }
}

// ---------------- K2/K3/K4: segment softmax + weighted scatter ---------------
__global__ void __launch_bounds__(256) k2_max(const int* __restrict__ src,
                                              const int* __restrict__ dst,
                                              const float* __restrict__ aNode,
                                              unsigned* __restrict__ amaxU) {
  unsigned g = blockIdx.x * 256u + threadIdx.x;
  if (g >= (unsigned)NE * 4u) return;
  unsigned e = g >> 2, h = g & 3;
  float a = aNode[src[e] * 4 + h];
  atomicMax(&amaxU[dst[e] * 4 + h], fenc(a));
}

__global__ void __launch_bounds__(256) k3_denom(const int* __restrict__ src,
                                                const int* __restrict__ dst,
                                                const float* __restrict__ aNode,
                                                const unsigned* __restrict__ amaxU,
                                                float* __restrict__ denom) {
  unsigned g = blockIdx.x * 256u + threadIdx.x;
  if (g >= (unsigned)NE * 4u) return;
  unsigned e = g >> 2, h = g & 3;
  int d = dst[e];
  float a = aNode[src[e] * 4 + h];
  unsafeAtomicAdd(&denom[d * 4 + h], __expf(a - fdec(amaxU[d * 4 + h])));
}

__global__ void __launch_bounds__(256) k4_scatter(const int* __restrict__ src,
                                                  const int* __restrict__ dst,
                                                  const float* __restrict__ aNode,
                                                  const unsigned* __restrict__ amaxU,
                                                  const float* __restrict__ denom,
                                                  const __bf16* __restrict__ xVb,
                                                  float* __restrict__ agg) {
  unsigned g = blockIdx.x * 256u + threadIdx.x;  // exactly NE*128 threads
  unsigned e = g >> 7, c = g & 127, h = c >> 5;
  int s = src[e], d = dst[e];
  float a = aNode[s * 4 + h];
  float w = __expf(a - fdec(amaxU[d * 4 + h])) / (denom[d * 4 + h] + 1e-16f);
  float v = (float)xVb[(size_t)s * FD + c];
  unsafeAtomicAdd(&agg[(size_t)d * FD + c], v * w);
}

// ---------------- K5: +att_r, LN0, rFF (2x WMMA GEMM), residual, LN1 ---------
#define XSTR 132  // fp32 row stride (528 B, 16B-aligned)
__global__ void __launch_bounds__(256) k5_ff(const float* __restrict__ agg,
                                             const float* __restrict__ att,
                                             const float* __restrict__ g0,
                                             const float* __restrict__ be0,
                                             const __bf16* __restrict__ w1T,  // bf16 W1^T
                                             const float* __restrict__ b1,
                                             const __bf16* __restrict__ w2T,  // bf16 W2^T
                                             const float* __restrict__ b2,
                                             const float* __restrict__ g1,
                                             const float* __restrict__ be1,
                                             float* __restrict__ out) {
  extern __shared__ unsigned char smem[];
  float*  sX = (float*)smem;                                  // [128][XSTR] f32 residual
  __bf16* sW = (__bf16*)(smem + 128 * XSTR * 4);              // [128][KTR] bf16 W^T
  __bf16* sA = (__bf16*)(smem + 128 * XSTR * 4 + WT_BYTES);   // [128][KTR] bf16 A-staging
  const int tid = threadIdx.x;
  const int base = blockIdx.x * 128;

  // async-DMA W1^T into sW; overlap with the agg block load + LN0
  for (int i = tid; i < WT_BYTES / 16; i += 256)
    lds_stage16((char*)sW + i * 16, (const char*)w1T + i * 16);

  for (int i = tid; i < 128 * 128; i += 256) {
    int r = i >> 7, c = i & 127;
    int row = base + r; if (row >= NHE) row = NHE - 1;
    sX[r * XSTR + c] = agg[(size_t)row * FD + c] + att[c];  // out + att_r
  }
  __syncthreads();
  if (tid < 128) {  // LN0, thread-per-row, float4 (b128) accesses
    float4* row = (float4*)(sX + tid * XSTR);
    float s = 0.f, ss = 0.f;
#pragma unroll
    for (int q = 0; q < 32; ++q) {
      float4 v = row[q];
      s += v.x + v.y + v.z + v.w;
      ss += v.x * v.x + v.y * v.y + v.z * v.z + v.w * v.w;
    }
    float mu = s * (1.f / 128.f);
    float rstd = rsqrtf(ss * (1.f / 128.f) - mu * mu + 1e-5f);
    __bf16* ar = sA + tid * KTR;
#pragma unroll
    for (int q = 0; q < 32; ++q) {
      float4 v = row[q];
      float4 g = ((const float4*)g0)[q];
      float4 bb = ((const float4*)be0)[q];
      v.x = (v.x - mu) * rstd * g.x + bb.x;
      v.y = (v.y - mu) * rstd * g.y + bb.y;
      v.z = (v.z - mu) * rstd * g.z + bb.z;
      v.w = (v.w - mu) * rstd * g.w + bb.w;
      row[q] = v;  // fp32 for residual
      ar[4 * q + 0] = (__bf16)v.x;  // bf16 A tile for GEMM1
      ar[4 * q + 1] = (__bf16)v.y;
      ar[4 * q + 2] = (__bf16)v.z;
      ar[4 * q + 3] = (__bf16)v.w;
    }
  }
  lds_stage_wait();
  __syncthreads();

  const int lane = tid & 31, wv = tid >> 5;
  const int lo = lane & 15, hi = lane >> 4;
  const int m = wv * 16 + lo;

  // GEMM1: A rows are wave-private in sA -> read-then-overwrite is safe (DS in-order)
  v16bf aF[4];
#pragma unroll
  for (int kc = 0; kc < 4; ++kc) aF[kc] = afrag_b(sA, KTR, m, kc, hi);
#pragma unroll
  for (int t = 0; t < 8; ++t) {
    float bb = b1[t * 16 + lo];
    v8f acc = {bb, bb, bb, bb, bb, bb, bb, bb};
#pragma unroll
    for (int kc = 0; kc < 4; ++kc)
      acc = wmma_bf(aF[kc], bfrag_t(sW, t * 16 + lo, kc, hi), acc);
#pragma unroll
    for (int j = 0; j < 8; ++j) {
      float v = acc[j]; v = v > 0.f ? v : 0.f;
      sA[(wv * 16 + j + 8 * hi) * KTR + t * 16 + lo] = (__bf16)v;
    }
  }
  __syncthreads();
  // async-DMA W2^T over sW; overlap with the A2 fragment loads from sA
  for (int i = tid; i < WT_BYTES / 16; i += 256)
    lds_stage16((char*)sW + i * 16, (const char*)w2T + i * 16);
  v16bf aF2[4];
#pragma unroll
  for (int kc = 0; kc < 4; ++kc) aF2[kc] = afrag_b(sA, KTR, m, kc, hi);
  lds_stage_wait();
  __syncthreads();

#pragma unroll
  for (int t = 0; t < 8; ++t) {  // GEMM2, residual y = x0 + relu(h2)
    float bb = b2[t * 16 + lo];
    v8f acc = {bb, bb, bb, bb, bb, bb, bb, bb};
#pragma unroll
    for (int kc = 0; kc < 4; ++kc)
      acc = wmma_bf(aF2[kc], bfrag_t(sW, t * 16 + lo, kc, hi), acc);
#pragma unroll
    for (int j = 0; j < 8; ++j) {
      float v = acc[j]; v = v > 0.f ? v : 0.f;
      sX[(wv * 16 + j + 8 * hi) * XSTR + t * 16 + lo] += v;
    }
  }
  __syncthreads();
  if (tid < 128) {  // LN1 + store, float4 accesses
    float4* row = (float4*)(sX + tid * XSTR);
    float s = 0.f, ss = 0.f;
#pragma unroll
    for (int q = 0; q < 32; ++q) {
      float4 v = row[q];
      s += v.x + v.y + v.z + v.w;
      ss += v.x * v.x + v.y * v.y + v.z * v.z + v.w * v.w;
    }
    float mu = s * (1.f / 128.f);
    float rstd = rsqrtf(ss * (1.f / 128.f) - mu * mu + 1e-5f);
    int orow = base + tid;
    if (orow < NHE) {
      float4* op = (float4*)(out + (size_t)orow * FD);
#pragma unroll
      for (int q = 0; q < 32; ++q) {
        float4 v = row[q];
        float4 g = ((const float4*)g1)[q];
        float4 bb = ((const float4*)be1)[q];
        v.x = (v.x - mu) * rstd * g.x + bb.x;
        v.y = (v.y - mu) * rstd * g.y + bb.y;
        v.z = (v.z - mu) * rstd * g.z + bb.z;
        v.w = (v.w - mu) * rstd * g.w + bb.w;
        op[q] = v;
      }
    }
  }
}

// ---------------- host launcher ----------------------------------------------
extern "C" void kernel_launch(void* const* d_in, const int* in_sizes, int n_in,
                              void* d_out, int out_size, void* d_ws, size_t ws_size,
                              hipStream_t stream) {
  (void)in_sizes; (void)n_in; (void)out_size; (void)ws_size;
  const float* x    = (const float*)d_in[0];
  const int*   src  = (const int*)d_in[1];
  const int*   dst  = (const int*)d_in[2];
  const float* W_K  = (const float*)d_in[3];
  const float* b_K  = (const float*)d_in[4];
  const float* W_V  = (const float*)d_in[5];
  const float* b_V  = (const float*)d_in[6];
  const float* attr = (const float*)d_in[7];
  const float* ln0g = (const float*)d_in[8];
  const float* ln0b = (const float*)d_in[9];
  const float* W1   = (const float*)d_in[10];
  const float* b1   = (const float*)d_in[11];
  const float* W2   = (const float*)d_in[12];
  const float* b2   = (const float*)d_in[13];
  const float* ln1g = (const float*)d_in[14];
  const float* ln1b = (const float*)d_in[15];
  float* out = (float*)d_out;

  char* ws = (char*)d_ws;
  __bf16*   xVb   = (__bf16*)(ws + 0);                 // 200000*128*2  = 51,200,000
  float*    aNode = (float*)(ws + 51200000);           // 200000*4*4    =  3,200,000
  unsigned* amaxU = (unsigned*)(ws + 54400000);        // 100000*4*4    =  1,600,000
  float*    denom = (float*)(ws + 56000000);           // 100000*4*4    =  1,600,000
  float*    agg   = (float*)(ws + 57600000);           // 100000*128*4  = 51,200,000
  __bf16*   wtAll = (__bf16*)(ws + 108800000);         // 4 * 34816     =    139,264

  const size_t k1_lds = 2 * WT_BYTES + 128 * 4;                    // ~70 KB
  const size_t k5_lds = 128 * XSTR * 4 + WT_BYTES + WT_BYTES;      // ~137 KB

  k_pre<<<256, 256, 0, stream>>>(W_K, W_V, W1, W2, wtAll);
  k0_init<<<(NHE * 128 + 255) / 256, 256, 0, stream>>>(amaxU, denom, agg);
  k1_proj<<<(NN + 127) / 128, 256, k1_lds, stream>>>(x, wtAll, b_K, b_V, attr, xVb, aNode);
  k2_max<<<(NE * 4 + 255) / 256, 256, 0, stream>>>(src, dst, aNode, amaxU);
  k3_denom<<<(NE * 4 + 255) / 256, 256, 0, stream>>>(src, dst, aNode, amaxU, denom);
  k4_scatter<<<(NE * 128) / 256, 256, 0, stream>>>(src, dst, aNode, amaxU, denom, xVb, agg);
  k5_ff<<<(NHE + 127) / 128, 256, k5_lds, stream>>>(agg, attr, ln0g, ln0b,
                                                    wtAll + 2 * (128 * KTR), b1,
                                                    wtAll + 3 * (128 * KTR), b2,
                                                    ln1g, ln1b, out);
}
